// GNNDecoder_63230508532132
// MI455X (gfx1250) — compile-verified, matrix operand order checked by reading, not run on hardware
//
#include <hip/hip_runtime.h>
#include <hip/hip_bf16.h>

// ---------------- problem constants ----------------
#define GN_N   100000
#define GN_E   1600000
#define GN_H   64
#define GN_L   5
#define BN_EPS 1e-5f

#define ETILES      (GN_E / 16)   // 100000 edge tiles
#define NTILES      (GN_N / 16)   // 6250 node tiles
#define EDGE_BLOCKS 1250          // 10000 waves -> 10 tiles/wave
#define NODE_BLOCKS 98            // 784 waves -> ~8 tiles/wave

typedef __attribute__((ext_vector_type(16))) _Float16 v16h;
typedef __attribute__((ext_vector_type(8)))  _Float16 v8h;
typedef __attribute__((ext_vector_type(8)))  float    v8f;
typedef __attribute__((ext_vector_type(4)))  float    v4f;
typedef __attribute__((ext_vector_type(4)))  int      v4i;

// ---------------- async LDS staging (CDNA5 path, guarded) ----------------
#if __has_builtin(__builtin_amdgcn_global_load_async_to_lds_b128) && \
    __has_builtin(__builtin_amdgcn_s_wait_asynccnt)
#define HAVE_ASYNC_LDS 1
#else
#define HAVE_ASYNC_LDS 0
#endif

__device__ __forceinline__ void stage_b128(const void* g, void* s) {
#if HAVE_ASYNC_LDS
    __builtin_amdgcn_global_load_async_to_lds_b128((v4i*)g, (v4i*)s, 0, 0);
#else
    *(v8h*)s = *(const v8h*)g;
#endif
}
__device__ __forceinline__ void stage_wait() {
#if HAVE_ASYNC_LDS
    __builtin_amdgcn_s_wait_asynccnt(0);
#endif
    __syncthreads();
}

// ---------------- WMMA helpers ----------------
// D = A(16x32 f16) * B(32x16 f16) + C(16x16 f32)
__device__ __forceinline__ v8f wmma16(v16h a, v16h b, v8f c) {
    return __builtin_amdgcn_wmma_f32_16x16x32_f16(
        /*neg_a=*/false, a, /*neg_b=*/false, b,
        /*c_mod=*/(short)0, c, /*reuse_a=*/false, /*reuse_b=*/false);
}

// v16h fragment for one 16x32 (MxK) k-tile from a row of halves.
// ISA 16-bit layout: lane {m = lane&15, half = lane>>4}; element i maps to
// K = koff + (i<8 ? 0 : 16) + 8*half + (i%8) -> two contiguous 8-half loads.
__device__ __forceinline__ v16h load_frag_h(const _Float16* __restrict__ row,
                                            int koff, int half) {
    v8h lo = *(const v8h*)(row + koff + 8 * half);
    v8h hi = *(const v8h*)(row + koff + 16 + 8 * half);
    return __builtin_shufflevector(lo, hi, 0,1,2,3,4,5,6,7,8,9,10,11,12,13,14,15);
}

// Same fragment from an f32 row (converted on the fly).
__device__ __forceinline__ v16h load_frag_f32(const float* __restrict__ row,
                                              int koff, int half) {
    v4f a = *(const v4f*)(row + koff + 8 * half);
    v4f b = *(const v4f*)(row + koff + 8 * half + 4);
    v4f c = *(const v4f*)(row + koff + 16 + 8 * half);
    v4f d = *(const v4f*)(row + koff + 16 + 8 * half + 4);
    v16h r;
#pragma unroll
    for (int i = 0; i < 4; ++i) {
        r[i]      = (_Float16)a[i];
        r[4 + i]  = (_Float16)b[i];
        r[8 + i]  = (_Float16)c[i];
        r[12 + i] = (_Float16)d[i];
    }
    return r;
}

// ---------------- kernel 1: pack weights to transposed f16 ----------------
__global__ __launch_bounds__(256) void k_pack_weights(
    const float* __restrict__ mw1, const float* __restrict__ mw2,
    const float* __restrict__ uw1, const float* __restrict__ uw2,
    _Float16* __restrict__ mw1t, _Float16* __restrict__ mw2t,
    _Float16* __restrict__ uw1t, _Float16* __restrict__ uw2t) {
    int i = blockIdx.x * 256 + threadIdx.x;
    if (i < GN_L * 64 * 64) {
        int l = i / 4096, r = i % 4096, k = r / 64, n = r % 64;
        int o = l * 4096 + n * 64 + k;
        mw1t[o] = (_Float16)mw1[i];
        mw2t[o] = (_Float16)mw2[i];
        uw2t[o] = (_Float16)uw2[i];
    }
    if (i < GN_L * 128 * 64) {
        int l = i / 8192, r = i % 8192, k = r / 64, n = r % 64;
        uw1t[l * 8192 + n * 128 + k] = (_Float16)uw1[i];
    }
}

// ---------------- kernel 2: input projection ----------------
__global__ __launch_bounds__(256) void k_input_proj(
    const float* __restrict__ x, const float* __restrict__ w,
    const float* __restrict__ b, float* __restrict__ h,
    _Float16* __restrict__ h16) {
    long i = (long)blockIdx.x * 256 + threadIdx.x;
    if (i >= (long)GN_N * GN_H) return;
    int node = (int)(i >> 6), c = (int)(i & 63);
    float v = b[c] + x[node * 3 + 0] * w[0 * 64 + c]
                   + x[node * 3 + 1] * w[1 * 64 + c]
                   + x[node * 3 + 2] * w[2 * 64 + c];
    v = v > 0.f ? v : 0.f;
    h[i]   = v;
    h16[i] = (_Float16)v;
}

// ---------------- kernel 3: zero agg ----------------
__global__ __launch_bounds__(256) void k_zero(v4f* __restrict__ p, long n4) {
    long i = (long)blockIdx.x * 256 + threadIdx.x;
    if (i < n4) p[i] = (v4f){0.f, 0.f, 0.f, 0.f};
}

// ---------------- kernel 4: edge MLP + scatter-add ----------------
// Persistent waves: each wave loops over edge tiles; all 16 B-fragments for
// both GEMMs live in registers for the whole loop (128 VGPRs).
__global__ __launch_bounds__(256) void k_edge(
    const _Float16* __restrict__ h16,
    const int* __restrict__ src, const int* __restrict__ dst,
    const _Float16* __restrict__ w1t, const _Float16* __restrict__ w2t,
    const float* __restrict__ b1, const float* __restrict__ b2,
    float* __restrict__ agg) {
    __shared__ __align__(16) _Float16 sW1[64 * 64];
    __shared__ __align__(16) _Float16 sW2[64 * 64];
    __shared__ __align__(16) _Float16 sTmp[8 * 16 * 64];

    int tid = threadIdx.x;
#pragma unroll
    for (int i = 0; i < 2; ++i) {   // async weight staging into LDS
        stage_b128((const v8h*)w1t + tid + 256 * i, (v8h*)sW1 + tid + 256 * i);
        stage_b128((const v8h*)w2t + tid + 256 * i, (v8h*)sW2 + tid + 256 * i);
    }
    stage_wait();

    int wave = tid >> 5, lane = tid & 31;
    int half = lane >> 4, mn = lane & 15;

    // hoist all B fragments into registers (loop-invariant)
    v16h B1[4][2], B2[4][2];
#pragma unroll
    for (int nt = 0; nt < 4; ++nt) {
        const _Float16* r1 = sW1 + (nt * 16 + mn) * 64;
        const _Float16* r2 = sW2 + (nt * 16 + mn) * 64;
#pragma unroll
        for (int kt = 0; kt < 2; ++kt) {
            B1[nt][kt] = load_frag_h(r1, 32 * kt, half);
            B2[nt][kt] = load_frag_h(r2, 32 * kt, half);
        }
    }
    float bias1[4], bias2[4];
#pragma unroll
    for (int nt = 0; nt < 4; ++nt) {
        bias1[nt] = b1[nt * 16 + mn];
        bias2[nt] = b2[nt * 16 + mn];
    }

    _Float16* tmp = sTmp + wave * (16 * 64);
    const int stride = EDGE_BLOCKS * 8;
    int gw = blockIdx.x * 8 + wave;

    for (int tile = gw; tile < ETILES; tile += stride) {  // wave-uniform
        int e0 = tile * 16;
        int srow = src[e0 + mn];
        const _Float16* arow = h16 + (long)srow * GN_H;
        v16h a0 = load_frag_h(arow, 0, half);
        v16h a1 = load_frag_h(arow, 32, half);

        if (tile + stride < ETILES) {  // prefetch next tile's gather rows
            int sn = src[(tile + stride) * 16 + mn];
            __builtin_prefetch(h16 + (long)sn * GN_H, 0, 3);
        }

        // GEMM1 + bias + relu -> LDS tile (C-layout -> row-major f16)
#pragma unroll
        for (int nt = 0; nt < 4; ++nt) {
            v8f acc = {};
            acc = wmma16(a0, B1[nt][0], acc);
            acc = wmma16(a1, B1[nt][1], acc);
            int n = nt * 16 + mn;
#pragma unroll
            for (int r = 0; r < 8; ++r) {
                float v = acc[r] + bias1[nt];
                v = v > 0.f ? v : 0.f;
                tmp[(r + 8 * half) * 64 + n] = (_Float16)v;
            }
        }
        // wave-private LDS round-trip: C layout -> A layout
        v16h t0 = load_frag_h(tmp + mn * 64, 0, half);
        v16h t1 = load_frag_h(tmp + mn * 64, 32, half);

        // GEMM2 + bias, scatter-add into agg[dst] (hits L2)
#pragma unroll
        for (int nt = 0; nt < 4; ++nt) {
            v8f acc = {};
            acc = wmma16(t0, B2[nt][0], acc);
            acc = wmma16(t1, B2[nt][1], acc);
            int n = nt * 16 + mn;
#pragma unroll
            for (int r = 0; r < 8; ++r) {
                int d = dst[e0 + r + 8 * half];
                atomicAdd(&agg[(long)d * GN_H + n], acc[r] + bias2[nt]);
            }
        }
    }
}

// ---------------- kernel 5: node update (concat GEMM + BN + residual) ----------------
__global__ __launch_bounds__(256) void k_node(
    float* __restrict__ h, _Float16* __restrict__ h16,
    const float* __restrict__ agg,
    const _Float16* __restrict__ w1t,  // [n][k], k=128
    const _Float16* __restrict__ w2t,  // [n][k], k=64
    const float* __restrict__ b1, const float* __restrict__ b2,
    const float* __restrict__ bng, const float* __restrict__ bnb,
    const float* __restrict__ bnm, const float* __restrict__ bnv) {
    __shared__ __align__(16) _Float16 sW1[64 * 128];
    __shared__ __align__(16) _Float16 sW2[64 * 64];
    __shared__ __align__(16) _Float16 sTmp[8 * 16 * 64];

    int tid = threadIdx.x;
#pragma unroll
    for (int i = 0; i < 4; ++i)
        stage_b128((const v8h*)w1t + tid + 256 * i, (v8h*)sW1 + tid + 256 * i);
#pragma unroll
    for (int i = 0; i < 2; ++i)
        stage_b128((const v8h*)w2t + tid + 256 * i, (v8h*)sW2 + tid + 256 * i);
    stage_wait();

    int wave = tid >> 5, lane = tid & 31;
    int half = lane >> 4, mn = lane & 15;
    _Float16* tmp = sTmp + wave * (16 * 64);
    const int stride = NODE_BLOCKS * 8;
    int gw = blockIdx.x * 8 + wave;

    for (int tile = gw; tile < NTILES; tile += stride) {  // wave-uniform
        int n0 = tile * 16;
        const _Float16* hr = h16 + (long)(n0 + mn) * GN_H;
        const float*    ar = agg + (long)(n0 + mn) * GN_H;
        v16h a0 = load_frag_h(hr, 0, half);
        v16h a1 = load_frag_h(hr, 32, half);
        v16h a2 = load_frag_f32(ar, 0, half);
        v16h a3 = load_frag_f32(ar, 32, half);

#pragma unroll
        for (int nt = 0; nt < 4; ++nt) {
            const _Float16* brow = sW1 + (nt * 16 + mn) * 128;
            v8f acc = {};
            acc = wmma16(a0, load_frag_h(brow, 0, half), acc);
            acc = wmma16(a1, load_frag_h(brow, 32, half), acc);
            acc = wmma16(a2, load_frag_h(brow, 64, half), acc);
            acc = wmma16(a3, load_frag_h(brow, 96, half), acc);
            int n = nt * 16 + mn;
            float bias = b1[n];
#pragma unroll
            for (int r = 0; r < 8; ++r) {
                float v = acc[r] + bias;
                v = v > 0.f ? v : 0.f;
                tmp[(r + 8 * half) * 64 + n] = (_Float16)v;
            }
        }
        v16h t0 = load_frag_h(tmp + mn * 64, 0, half);
        v16h t1 = load_frag_h(tmp + mn * 64, 32, half);

#pragma unroll
        for (int nt = 0; nt < 4; ++nt) {
            const _Float16* brow = sW2 + (nt * 16 + mn) * 64;
            v8f acc = {};
            acc = wmma16(t0, load_frag_h(brow, 0, half), acc);
            acc = wmma16(t1, load_frag_h(brow, 32, half), acc);
            int n = nt * 16 + mn;
            float bias = b2[n];
            float mm = bnm[n], gg = bng[n], bb = bnb[n];
            float inv = rsqrtf(bnv[n] + BN_EPS);
#pragma unroll
            for (int r = 0; r < 8; ++r) {
                long row = n0 + r + 8 * half;
                float hn = acc[r] + bias;
                hn = (hn - mm) * inv * gg + bb;
                float out = hn + h[row * GN_H + n];
                out = out > 0.f ? out : 0.f;
                h[row * GN_H + n]   = out;          // in-place: wave owns rows
                h16[row * GN_H + n] = (_Float16)out;
            }
        }
    }
}

// ---------------- kernel 6: output head ----------------
__global__ __launch_bounds__(256) void k_head(
    const float* __restrict__ h, const float* __restrict__ w1,
    const float* __restrict__ b1, const float* __restrict__ w2,
    const float* __restrict__ b2, float* __restrict__ out, int nq) {
    __shared__ float sw1[64 * 32];
    __shared__ float sw2[32];
    __shared__ float sb1[32];
    int tid = threadIdx.x;
#pragma unroll
    for (int i = 0; i < 8; ++i) sw1[tid + 256 * i] = w1[tid + 256 * i];
    if (tid < 32) { sw2[tid] = w2[tid]; sb1[tid] = b1[tid]; }
    __syncthreads();

    int i = blockIdx.x * 256 + tid;
    if (i >= nq) return;
    const float* r = h + (long)i * GN_H;
    float acc = b2[0];
#pragma unroll 4
    for (int j = 0; j < 32; ++j) {
        float s = sb1[j];
#pragma unroll
        for (int k = 0; k < 64; ++k) s += r[k] * sw1[k * 32 + j];
        s = s > 0.f ? s : 0.f;
        acc += s * sw2[j];
    }
    out[i] = acc;
}

// ---------------- launch ----------------
extern "C" void kernel_launch(void* const* d_in, const int* in_sizes, int n_in,
                              void* d_out, int out_size, void* d_ws, size_t ws_size,
                              hipStream_t stream) {
    const float* x      = (const float*)d_in[0];
    const int*   eidx   = (const int*)d_in[1];
    const float* w_in   = (const float*)d_in[3];
    const float* b_in   = (const float*)d_in[4];
    const float* msg_w1 = (const float*)d_in[5];
    const float* msg_b1 = (const float*)d_in[6];
    const float* msg_w2 = (const float*)d_in[7];
    const float* msg_b2 = (const float*)d_in[8];
    const float* upd_w1 = (const float*)d_in[9];
    const float* upd_b1 = (const float*)d_in[10];
    const float* upd_w2 = (const float*)d_in[11];
    const float* upd_b2 = (const float*)d_in[12];
    const float* bn_g   = (const float*)d_in[13];
    const float* bn_b   = (const float*)d_in[14];
    const float* bn_m   = (const float*)d_in[15];
    const float* bn_v   = (const float*)d_in[16];
    const float* out_w1 = (const float*)d_in[17];
    const float* out_b1 = (const float*)d_in[18];
    const float* out_w2 = (const float*)d_in[19];
    const float* out_b2 = (const float*)d_in[20];

    const int* src = eidx;
    const int* dst = eidx + GN_E;

    // workspace layout (256B aligned)
    char* ws = (char*)d_ws;
    float*     h    = (float*)(ws + 0);                         // 25.6 MB
    _Float16*  h16  = (_Float16*)(ws + 25600000);               // 12.8 MB
    float*     agg  = (float*)(ws + 38400000);                  // 25.6 MB
    _Float16*  mw1t = (_Float16*)(ws + 64000000);
    _Float16*  mw2t = (_Float16*)(ws + 64040960);
    _Float16*  uw1t = (_Float16*)(ws + 64081920);
    _Float16*  uw2t = (_Float16*)(ws + 64163840);

    k_pack_weights<<<(GN_L * 8192 + 255) / 256, 256, 0, stream>>>(
        msg_w1, msg_w2, upd_w1, upd_w2, mw1t, mw2t, uw1t, uw2t);

    k_input_proj<<<(GN_N * GN_H + 255) / 256, 256, 0, stream>>>(
        x, w_in, b_in, h, h16);

    const long agg4 = (long)GN_N * GN_H / 4;

    for (int l = 0; l < GN_L; ++l) {
        k_zero<<<(int)((agg4 + 255) / 256), 256, 0, stream>>>((v4f*)agg, agg4);
        k_edge<<<EDGE_BLOCKS, 256, 0, stream>>>(
            h16, src, dst, mw1t + l * 4096, mw2t + l * 4096,
            msg_b1 + l * 64, msg_b2 + l * 64, agg);
        k_node<<<NODE_BLOCKS, 256, 0, stream>>>(
            h, h16, agg, uw1t + l * 8192, uw2t + l * 4096,
            upd_b1 + l * 64, upd_b2 + l * 64,
            bn_g + l * 64, bn_b + l * 64, bn_m + l * 64, bn_v + l * 64);
    }

    k_head<<<(out_size + 255) / 256, 256, 0, stream>>>(
        h, out_w1, out_b1, out_w2, out_b2, (float*)d_out, out_size);
}
